// PositionAttention_91182155694507
// MI455X (gfx1250) — matrix operand order repaired and checked by reading, not compiled, not run on hardware
//
#include <hip/hip_runtime.h>
#include <math.h>

typedef __attribute__((ext_vector_type(16))) _Float16 v16h;
typedef __attribute__((ext_vector_type(8)))  float    v8f;

constexpr int kB  = 4;
constexpr int kC  = 512;
constexpr int kN  = 4096;   // 64*64
constexpr int kCq = 64;

union Frag {
    v16h h;
    _Float16 e[16];
    unsigned int u[8];
};

// ---------------------------------------------------------------------------
// Stage 1: projection GEMM  out[O,N] = W[O,C] @ x[C,N]  (per batch), f16 out.
// One wave per 16x16 output tile; K-loop of 16 x v_wmma_f32_16x16x32_f16.
// transposeOut=1 stores out as [N,O] (for K^T, feeding WMMA A-layout later).
// ---------------------------------------------------------------------------
__global__ __launch_bounds__(256)
void proj_kernel(const float* __restrict__ Wm, const float* __restrict__ x,
                 _Float16* __restrict__ out, int O, int transposeOut)
{
    const int lane = threadIdx.x & 31;
    const int ln   = lane & 15;        // A-row / B-col / D-col within tile
    const int hi   = lane >> 4;
    const int wid  = (int)((blockIdx.x * blockDim.x + threadIdx.x) >> 5);

    const int tilesPerBatch = (O / 16) * (kN / 16);
    const int b  = wid / tilesPerBatch;
    const int t  = wid % tilesPerBatch;
    const int obase = (t / (kN / 16)) * 16;
    const int nbase = (t % (kN / 16)) * 16;

    const float* xb = x + (size_t)b * kC * kN;

    v8f acc = {};
    for (int c0 = 0; c0 < kC; c0 += 32) {
        // A = W tile (16 rows x 32 K), f32 -> f16, contiguous K-pairs
        Frag A;
        const float* wrow = Wm + (size_t)(obase + ln) * kC + c0;
#pragma unroll
        for (int v = 0; v < 8; ++v) {
            int k0 = (hi << 4) + (v << 1);
            float2 w2 = *(const float2*)(wrow + k0);
            A.e[2 * v]     = (_Float16)w2.x;
            A.e[2 * v + 1] = (_Float16)w2.y;
        }
        // B = x tile (32 K x 16 cols), strided by N
        Frag Bf;
#pragma unroll
        for (int v = 0; v < 8; ++v) {
            int c = c0 + (hi << 4) + (v << 1);
            Bf.e[2 * v]     = (_Float16)xb[(size_t)c * kN + nbase + ln];
            Bf.e[2 * v + 1] = (_Float16)xb[(size_t)(c + 1) * kN + nbase + ln];
        }
        acc = __builtin_amdgcn_wmma_f32_16x16x32_f16(false, A.h, false, Bf.h,
                                                     (short)0, acc, false, false);
    }

    if (!transposeOut) {
        _Float16* ob = out + (size_t)b * O * kN;
#pragma unroll
        for (int r = 0; r < 8; ++r)
            ob[(size_t)(obase + r + 8 * hi) * kN + nbase + ln] = (_Float16)acc[r];
    } else {
        _Float16* ob = out + (size_t)b * kN * O;
#pragma unroll
        for (int r = 0; r < 8; ++r)
            ob[(size_t)(nbase + ln) * O + obase + r + 8 * hi] = (_Float16)acc[r];
    }
}

// ---------------------------------------------------------------------------
// Stage 2: flash attention, transposed-score formulation.
//   St[key,query] = K^T(A) x Q(B)   -> queries live in lanes of C/D layout
//   online softmax: in-lane reductions + one shfl_xor(16) half-swap
//   O[c,query]   += V(A) x P^T(B)   -> P^T B-fragment built via xor-16 swap
// Block = 8 waves: one 16-query tile, each wave owns 64 output channels.
// ---------------------------------------------------------------------------
__global__ __launch_bounds__(256)
void attn_kernel(const _Float16* __restrict__ qB,   // [B][Cq][N]
                 const _Float16* __restrict__ kT,   // [B][N][Cq]
                 const _Float16* __restrict__ vF,   // [B][C][N]
                 const float* __restrict__ x,
                 const float* __restrict__ gamma,
                 float* __restrict__ out)
{
    const int lane = threadIdx.x & 31;
    const int q    = lane & 15;         // query within tile (lanes) / A-row
    const int hi   = lane >> 4;
    const int wave = threadIdx.x >> 5;  // 0..7 -> channel group

    const int blk   = blockIdx.x;       // B * N/16 blocks
    const int b     = blk / (kN / 16);
    const int ibase = (blk % (kN / 16)) * 16;
    const int cbase = wave * 64;

    const _Float16* qb = qB + (size_t)b * kCq * kN;
    const _Float16* kb = kT + (size_t)b * kN * kCq;
    const _Float16* vb = vF + (size_t)b * kC * kN;

    // Q as B-matrix fragments (c-chunks 0..31, 32..63), loaded once.
    Frag Qf[2];
#pragma unroll
    for (int ch = 0; ch < 2; ++ch)
#pragma unroll
        for (int v = 0; v < 8; ++v) {
            int c = ch * 32 + (hi << 4) + (v << 1);
            Qf[ch].e[2 * v]     = qb[(size_t)c * kN + ibase + q];
            Qf[ch].e[2 * v + 1] = qb[(size_t)(c + 1) * kN + ibase + q];
        }

    float mrun = -3.0e38f, lrun = 0.0f;
    v8f acc[4] = {{}, {}, {}, {}};

    for (int jb = 0; jb < kN; jb += 32) {
        // ---- energy: St[key,query] for keys jb..jb+31 (two 16-key subtiles)
        v8f st[2];
#pragma unroll
        for (int sub = 0; sub < 2; ++sub) {
            v8f s = {};
#pragma unroll
            for (int ch = 0; ch < 2; ++ch) {
                Frag Ka;   // A-row = key, K-dim = c, contiguous pairs in kT
                const _Float16* krow =
                    kb + (size_t)(jb + sub * 16 + q) * kCq + ch * 32;
#pragma unroll
                for (int v = 0; v < 8; ++v) {
                    int k0 = (hi << 4) + (v << 1);
                    Ka.u[v] = *(const unsigned int*)(krow + k0);
                }
                s = __builtin_amdgcn_wmma_f32_16x16x32_f16(
                        false, Ka.h, false, Qf[ch].h, (short)0, s, false, false);
            }
            st[sub] = s;
        }

        // ---- online softmax over the 32 keys (per lane = per query)
        float tmax = st[0][0];
#pragma unroll
        for (int r = 0; r < 8; ++r) {
            tmax = fmaxf(tmax, st[0][r]);
            tmax = fmaxf(tmax, st[1][r]);
        }
        tmax = fmaxf(tmax, __shfl_xor(tmax, 16, 32));
        float mnew  = fmaxf(mrun, tmax);
        float alpha = __expf(mrun - mnew);

        float p1[8], p2[8], ssum = 0.0f;
#pragma unroll
        for (int r = 0; r < 8; ++r) {
            p1[r] = __expf(st[0][r] - mnew);
            p2[r] = __expf(st[1][r] - mnew);
            ssum += p1[r] + p2[r];
        }
        ssum += __shfl_xor(ssum, 16, 32);
        lrun = lrun * alpha + ssum;
        mrun = mnew;

        // ---- build P^T as B-matrix fragment via one half-swap
        float o1[8], o2[8];
#pragma unroll
        for (int r = 0; r < 8; ++r) {
            o1[r] = __shfl_xor(p1[r], 16, 32);
            o2[r] = __shfl_xor(p2[r], 16, 32);
        }
        Frag Pf;
#pragma unroll
        for (int v = 0; v < 4; ++v) {   // j = hi*16 + 2v, 2v in [0,8)
            float a0 = hi ? o2[2 * v]     : p1[2 * v];
            float a1 = hi ? o2[2 * v + 1] : p1[2 * v + 1];
            Pf.e[2 * v]     = (_Float16)a0;
            Pf.e[2 * v + 1] = (_Float16)a1;
        }
#pragma unroll
        for (int v = 4; v < 8; ++v) {   // 2v in [8,16)
            float a0 = hi ? p2[2 * v - 8] : o1[2 * v - 8];
            float a1 = hi ? p2[2 * v - 7] : o1[2 * v - 7];
            Pf.e[2 * v]     = (_Float16)a0;
            Pf.e[2 * v + 1] = (_Float16)a1;
        }

        // ---- rescale running accumulators
#pragma unroll
        for (int cc = 0; cc < 4; ++cc)
#pragma unroll
            for (int r = 0; r < 8; ++r) acc[cc][r] *= alpha;

        // ---- O[c,query] += V x P^T  (4 channel sub-tiles, K=32 in one WMMA)
#pragma unroll
        for (int cc = 0; cc < 4; ++cc) {
            Frag Va;   // A-row = channel, K-dim = j, contiguous pairs in vF
            const _Float16* vrow =
                vb + (size_t)(cbase + cc * 16 + q) * kN + jb;
#pragma unroll
            for (int v = 0; v < 8; ++v) {
                int k0 = (hi << 4) + (v << 1);
                Va.u[v] = *(const unsigned int*)(vrow + k0);
            }
            acc[cc] = __builtin_amdgcn_wmma_f32_16x16x32_f16(
                          false, Va.h, false, Pf.h, (short)0, acc[cc], false, false);
        }
    }

    // ---- epilogue: out = gamma * (acc / l) + x
    const float g    = gamma[0];
    const float linv = 1.0f / lrun;
    const float* xb  = x   + (size_t)b * kC * kN;
    float*       ob  = out + (size_t)b * kC * kN;
#pragma unroll
    for (int cc = 0; cc < 4; ++cc)
#pragma unroll
        for (int r = 0; r < 8; ++r) {
            int c = cbase + cc * 16 + r + 8 * hi;
            int i = ibase + q;
            ob[(size_t)c * kN + i] = g * acc[cc][r] * linv + xb[(size_t)c * kN + i];
        }
}

// ---------------------------------------------------------------------------
extern "C" void kernel_launch(void* const* d_in, const int* in_sizes, int n_in,
                              void* d_out, int out_size, void* d_ws, size_t ws_size,
                              hipStream_t stream)
{
    const float* x     = (const float*)d_in[0];
    const float* Wq    = (const float*)d_in[1];
    const float* Wk    = (const float*)d_in[2];
    const float* Wv    = (const float*)d_in[3];
    const float* gamma = (const float*)d_in[4];
    float* out = (float*)d_out;

    // workspace: f16 Q [B][Cq][N], K^T [B][N][Cq], V [B][C][N]  (20 MB total)
    _Float16* qB = (_Float16*)d_ws;
    _Float16* kT = qB + (size_t)kB * kCq * kN;
    _Float16* vF = kT + (size_t)kB * kN * kCq;

    const int tilesQK = kB * (kCq / 16) * (kN / 16);   // 4096 tiles
    const int tilesV  = kB * (kC  / 16) * (kN / 16);   // 32768 tiles

    proj_kernel<<<tilesQK / 8, 256, 0, stream>>>(Wq, x, qB, kCq, 0);
    proj_kernel<<<tilesQK / 8, 256, 0, stream>>>(Wk, x, kT, kCq, 1);
    proj_kernel<<<tilesV  / 8, 256, 0, stream>>>(Wv, x, vF, kC,  0);

    attn_kernel<<<kB * (kN / 16), 256, 0, stream>>>(qB, kT, vF, x, gamma, out);
}
